// TR2_49555332661799
// MI455X (gfx1250) — matrix operand order repaired and checked by reading, not compiled, not run on hardware
//
#include <hip/hip_runtime.h>

// ---------------------------------------------------------------------------
// Problem constants
// ---------------------------------------------------------------------------
#define N_REL   65536
#define D_FEAT  3872
#define D_OUT   512
#define N_CLS   36

typedef __attribute__((ext_vector_type(16))) __bf16 v16bf;
typedef __attribute__((ext_vector_type(8)))  __bf16 v8bf;
typedef __attribute__((ext_vector_type(4)))  __bf16 v4bf;
typedef __attribute__((ext_vector_type(8)))  float  v8f;

// ---------------------------------------------------------------------------
// Stage 1: per-block label histograms (64 blocks x 1024 items)
// ---------------------------------------------------------------------------
__global__ void hist_kernel(const int* __restrict__ label, int* __restrict__ blockHist) {
    __shared__ int h[N_CLS];
    int t = threadIdx.x;                       // 256 threads
    if (t < N_CLS) h[t] = 0;
    __syncthreads();
    int base = blockIdx.x * 1024 + t * 4;
    #pragma unroll
    for (int j = 0; j < 4; ++j) atomicAdd(&h[label[base + j]], 1);
    __syncthreads();
    if (t < N_CLS) blockHist[blockIdx.x * N_CLS + t] = h[t];
}

// ---------------------------------------------------------------------------
// Stage 2: serial exclusive prefix, class-major then block-minor.
// Also emits class_start/count and pair_base (cumsum of max(count-1,0)).
// ---------------------------------------------------------------------------
__global__ void prefix_kernel(const int* __restrict__ blockHist,
                              int* __restrict__ blockBase,
                              int* __restrict__ class_start,
                              int* __restrict__ class_count,
                              int* __restrict__ pair_base) {
    if (threadIdx.x != 0 || blockIdx.x != 0) return;
    int running = 0, pb = 0;
    for (int c = 0; c < N_CLS; ++c) {
        class_start[c] = running;
        for (int b = 0; b < 64; ++b) {
            blockBase[b * N_CLS + c] = running;
            running += blockHist[b * N_CLS + c];
        }
        int cc = running - class_start[c];
        class_count[c] = cc;
        pair_base[c] = pb;
        if (cc > 0) pb += cc - 1;
    }
}

// ---------------------------------------------------------------------------
// Stage 3: stable scatter. 64 blocks x 32 lanes; lane t owns 32 consecutive
// items, preserving original order -> exact stable counting sort.
// ---------------------------------------------------------------------------
__global__ void scatter_kernel(const int* __restrict__ label,
                               const int* __restrict__ blockBase,
                               int* __restrict__ order) {
    __shared__ int cnt[N_CLS][33];
    __shared__ int run[N_CLS][33];
    int t = threadIdx.x;                       // 32 threads
    int b = blockIdx.x;
    int base = b * 1024 + t * 32;
    for (int idx = t; idx < N_CLS * 33; idx += 32) (&cnt[0][0])[idx] = 0;
    __syncthreads();
    for (int j = 0; j < 32; ++j) cnt[label[base + j]][t]++;
    __syncthreads();
    for (int c = t; c < N_CLS; c += 32) {
        int s = blockBase[b * N_CLS + c];
        for (int tt = 0; tt < 32; ++tt) { run[c][tt] = s; s += cnt[c][tt]; }
    }
    __syncthreads();
    for (int j = 0; j < 32; ++j) {
        int i = base + j;
        int c = label[i];
        int p = run[c][t]++;                   // own column, no race
        order[p] = i;
    }
}

// ---------------------------------------------------------------------------
// Stage 4: emit (idx_lo, idx_hi) pairs for intra-class adjacent sorted rows.
// ---------------------------------------------------------------------------
__global__ void pairs_kernel(const int* __restrict__ order,
                             const int* __restrict__ label,
                             const int* __restrict__ class_start,
                             const int* __restrict__ class_count,
                             const int* __restrict__ pair_base,
                             int* __restrict__ idx_lo, int* __restrict__ idx_hi) {
    int k = blockIdx.x * 256 + threadIdx.x;
    if (k >= N_REL) return;
    int r = order[k];
    int c = label[r];
    int j = k - class_start[c];
    if (j < class_count[c] - 1) {
        int p = pair_base[c] + j;
        idx_lo[p] = r;
        idx_hi[p] = order[k + 1];
    }
}

// ---------------------------------------------------------------------------
// Weight repack: W[mat][k][n] fp32  ->  Wt[mat][n][k] bf16 (K-contiguous so a
// B fragment is two contiguous global_load_b128 per lane; 12MB -> L2 resident)
// ---------------------------------------------------------------------------
__global__ void wprep_kernel(const float* __restrict__ Wa,
                             const float* __restrict__ Ws,
                             const float* __restrict__ Wc,
                             __bf16* __restrict__ Wt) {
    __shared__ float tile[32][33];
    int mat = blockIdx.z;
    const float* W = (mat == 0) ? Wa : ((mat == 1) ? Ws : Wc);
    int k0 = blockIdx.x * 32, n0 = blockIdx.y * 32;
    int tx = threadIdx.x, ty = threadIdx.y;    // 32 x 8
    #pragma unroll
    for (int i = 0; i < 4; ++i)
        tile[ty + 8 * i][tx] = W[(size_t)(k0 + ty + 8 * i) * D_OUT + n0 + tx];
    __syncthreads();
    #pragma unroll
    for (int i = 0; i < 4; ++i)
        Wt[((size_t)mat * D_OUT + n0 + ty + 8 * i) * D_FEAT + k0 + tx] =
            (__bf16)tile[tx][ty + 8 * i];
}

// ---------------------------------------------------------------------------
// Fused diff + triple-GEMM.  BM=128, BN=64, BK=32, 8 waves (wave32).
// A tile (row diffs, bf16) staged in LDS and shared by all 3 weight matrices;
// B fragments streamed from L2-resident bf16 Wt.  12 v_wmma per K-step/wave.
// ---------------------------------------------------------------------------
__global__ __launch_bounds__(256)
void gemm_kernel(const float* __restrict__ G,
                 const __bf16* __restrict__ Wt,
                 const int* __restrict__ idx_lo,
                 const int* __restrict__ idx_hi,
                 const float* __restrict__ ba,
                 const float* __restrict__ bs,
                 const float* __restrict__ bc,
                 float* __restrict__ out, int M) {
    __shared__ __bf16 As[128][40];             // stride 40 elems = 80B (16B aligned)

    const int tid  = threadIdx.x;
    const int wave = tid >> 5, lane = tid & 31;
    const int wr   = wave >> 1;                // wave row   0..3  (32 rows each)
    const int wc   = wave & 1;                 // wave col   0..1  (32 cols each)
    const int lmod = lane & 15;
    const int lhalf = lane >> 4;

    const int m0  = blockIdx.x * 128;
    const int nb0 = blockIdx.y * 64;

    // Preload this thread's 4 pair indices (rows it stages each K-step).
    const int srow  = tid >> 3;                // 0..31
    const int scol4 = tid & 7;                 // 0..7  (float4 column)
    int ilo[4], ihi[4];
    #pragma unroll
    for (int i = 0; i < 4; ++i) {
        int rg = m0 + i * 32 + srow;
        bool v = rg < M;
        ilo[i] = v ? idx_lo[rg] : 0;
        ihi[i] = v ? idx_hi[rg] : 0;           // diff = 0 for padding rows
    }

    v8f acc[3][2][2] = {};

    for (int kb = 0; kb < D_FEAT; kb += 32) {
        // ---- stage A tile: diff(fp32) -> bf16 LDS -------------------------
        #pragma unroll
        for (int i = 0; i < 4; ++i) {
            const float4 lo = *(const float4*)(G + (size_t)ilo[i] * D_FEAT + kb + scol4 * 4);
            const float4 hi = *(const float4*)(G + (size_t)ihi[i] * D_FEAT + kb + scol4 * 4);
            v4bf d;
            d[0] = (__bf16)(hi.x - lo.x);
            d[1] = (__bf16)(hi.y - lo.y);
            d[2] = (__bf16)(hi.z - lo.z);
            d[3] = (__bf16)(hi.w - lo.w);
            *(v4bf*)&As[i * 32 + srow][scol4 * 4] = d;
        }
        __syncthreads();

        // ---- A fragments (ISA 16-bit A layout: lanes0-15 K0..7,16..23) ----
        v16bf af[2];
        #pragma unroll
        for (int tr = 0; tr < 2; ++tr) {
            const __bf16* p = &As[wr * 32 + tr * 16 + lmod][lhalf * 8];
            v8bf a0 = *(const v8bf*)p;
            v8bf a1 = *(const v8bf*)(p + 16);
            af[tr] = __builtin_shufflevector(a0, a1,
                         0,1,2,3,4,5,6,7,8,9,10,11,12,13,14,15);
        }

        // ---- B fragments straight from bf16 Wt (K-contiguous) + WMMAs -----
        #pragma unroll
        for (int mat = 0; mat < 3; ++mat) {
            #pragma unroll
            for (int tc = 0; tc < 2; ++tc) {
                const __bf16* bp = Wt
                    + ((size_t)mat * D_OUT + nb0 + wc * 32 + tc * 16 + lmod) * D_FEAT
                    + kb + lhalf * 8;
                v8bf b0 = *(const v8bf*)bp;
                v8bf b1 = *(const v8bf*)(bp + 16);
                v16bf bf = __builtin_shufflevector(b0, b1,
                               0,1,2,3,4,5,6,7,8,9,10,11,12,13,14,15);
                #pragma unroll
                for (int tr = 0; tr < 2; ++tr) {
                    acc[mat][tr][tc] = __builtin_amdgcn_wmma_f32_16x16x32_bf16(
                        false, af[tr], false, bf, (short)0, acc[mat][tr][tc],
                        false, false);
                }
            }
        }
        __syncthreads();
    }

    // ---- epilogue: bias + store (C/D layout: VGPR v -> M = v + 8*lhalf) ----
    const float* biases[3] = { ba, bs, bc };
    #pragma unroll
    for (int mat = 0; mat < 3; ++mat) {
        #pragma unroll
        for (int tc = 0; tc < 2; ++tc) {
            int col = nb0 + wc * 32 + tc * 16 + lmod;
            float bv = biases[mat][col];
            #pragma unroll
            for (int tr = 0; tr < 2; ++tr) {
                #pragma unroll
                for (int v = 0; v < 8; ++v) {
                    int row = m0 + wr * 32 + tr * 16 + v + 8 * lhalf;
                    if (row < M)
                        out[(size_t)mat * M * D_OUT + (size_t)row * D_OUT + col] =
                            acc[mat][tr][tc][v] + bv;
                }
            }
        }
    }
}

// ---------------------------------------------------------------------------
// Host-side launcher
// ---------------------------------------------------------------------------
extern "C" void kernel_launch(void* const* d_in, const int* in_sizes, int n_in,
                              void* d_out, int out_size, void* d_ws, size_t ws_size,
                              hipStream_t stream) {
    const float* G     = (const float*)d_in[0];
    const float* Wa    = (const float*)d_in[1];
    const float* ba    = (const float*)d_in[2];
    const float* Ws    = (const float*)d_in[3];
    const float* bs    = (const float*)d_in[4];
    const float* Wc    = (const float*)d_in[5];
    const float* bc    = (const float*)d_in[6];
    const int*   label = (const int*)d_in[7];
    float* out = (float*)d_out;

    const int M = out_size / (3 * D_OUT);      // harness-derived pair count

    // Workspace carving (~12.8 MB total)
    char* w = (char*)d_ws;
    int* order       = (int*)w;  w += (size_t)N_REL * 4;
    int* blockHist   = (int*)w;  w += 64 * N_CLS * 4;
    int* blockBase   = (int*)w;  w += 64 * N_CLS * 4;
    int* class_start = (int*)w;  w += N_CLS * 4;
    int* class_count = (int*)w;  w += N_CLS * 4;
    int* pair_base   = (int*)w;  w += N_CLS * 4;
    int* idx_lo      = (int*)w;  w += (size_t)N_REL * 4;
    int* idx_hi      = (int*)w;  w += (size_t)N_REL * 4;
    w = (char*)(((uintptr_t)w + 15) & ~(uintptr_t)15);
    __bf16* Wt       = (__bf16*)w;             // 3 * 512 * 3872 * 2 B

    hist_kernel   <<<64, 256, 0, stream>>>(label, blockHist);
    prefix_kernel <<<1, 32, 0, stream>>>(blockHist, blockBase, class_start,
                                         class_count, pair_base);
    scatter_kernel<<<64, 32, 0, stream>>>(label, blockBase, order);
    pairs_kernel  <<<N_REL / 256, 256, 0, stream>>>(order, label, class_start,
                                                    class_count, pair_base,
                                                    idx_lo, idx_hi);
    wprep_kernel  <<<dim3(D_FEAT / 32, D_OUT / 32, 3), dim3(32, 8), 0, stream>>>(
                      Wa, Ws, Wc, Wt);

    dim3 grid((M + 127) / 128 > 0 ? (M + 127) / 128 : 1, D_OUT / 64);
    gemm_kernel<<<grid, 256, 0, stream>>>(G, Wt, idx_lo, idx_hi,
                                          ba, bs, bc, out, M);
}
